// Attention_27977416966728
// MI455X (gfx1250) — matrix operand order repaired and checked by reading, not compiled, not run on hardware
//
#include <hip/hip_runtime.h>

// Problem constants (from reference): B=8, S=2048, E=1024, D=64
constexpr int Bn = 8;
constexpr int Sn = 2048;
constexpr int En = 1024;
constexpr int Dn = 64;

typedef __attribute__((ext_vector_type(16))) __bf16 v16bf;
typedef __attribute__((ext_vector_type(2)))  __bf16 v2bf;
typedef __attribute__((ext_vector_type(8)))  float  v8f;

union Frag16 { unsigned int u[8]; v16bf v; };

#if defined(__has_builtin)
#  if __has_builtin(__builtin_amdgcn_cvt_pk_bf16_f32)
#    define HAVE_CVT_PK 1
#  endif
#  if __has_builtin(__builtin_amdgcn_perm)
#    define HAVE_PERM 1
#  endif
#  if __has_builtin(__builtin_amdgcn_tensor_load_to_lds) && \
      __has_builtin(__builtin_amdgcn_s_wait_tensorcnt)
#    define HAVE_TDM 1
#  endif
#endif

// Pack two f32 into one dword of bf16x2 with round-to-nearest-even.
static __device__ __forceinline__ unsigned int pack_bf16x2(float lo, float hi) {
#ifdef HAVE_CVT_PK
    union { v2bf b; unsigned int u; } c;
    c.b = __builtin_amdgcn_cvt_pk_bf16_f32(lo, hi);
    return c.u;
#else
    union { float f; unsigned int u; } a, b;
    a.f = lo; b.f = hi;
    const unsigned int ra = a.u + 0x7FFFu + ((a.u >> 16) & 1u);
    const unsigned int rb = b.u + 0x7FFFu + ((b.u >> 16) & 1u);
#  ifdef HAVE_PERM
    return __builtin_amdgcn_perm(rb, ra, 0x07060302u);  // {rb[31:16], ra[31:16]}
#  else
    return (ra >> 16) | (rb & 0xFFFF0000u);
#  endif
#endif
}

static __device__ __forceinline__ unsigned short f2bf(float f) {
    return (unsigned short)(pack_bf16x2(f, 0.0f) & 0xFFFFu);
}

// 16-bit A-matrix 16x32 (MxK) per-lane K offset: lanes 0-15 hold K=0..15, lanes 16-31 hold K=16..31
static __device__ __forceinline__ int a_koff(int v, int half) {
    return 16 * half + 2 * v;
}
// 16-bit B-matrix 32x16 (KxN): lanes 0-15 hold K=0..15, lanes 16-31 hold K=16..31
static __device__ __forceinline__ int b_koff(int v, int half) {
    return 16 * half + 2 * v;
}

#ifdef HAVE_TDM
typedef __attribute__((ext_vector_type(4))) unsigned int v4u;
typedef __attribute__((ext_vector_type(8))) int         v8i;
typedef __attribute__((ext_vector_type(4))) int         v4i;

// Issue a TDM 2-D tile load (Global -> LDS). Descriptor layout per
// CDNA5 ISA.
//   tile_d0 elements contiguous (dim0), tile_d1 lines with stride_d0 elements,
//   data_size_code: 0=1B 1=2B 2=4B 3=8B.
// This toolchain's builtin takes 6 args: (g0, g1, g2, g3, g4, cpol).
static __device__ __forceinline__ void tdm_load_2d(unsigned int lds_off,
                                                   const void* gaddr,
                                                   unsigned int tile_d0,
                                                   unsigned int tile_d1,
                                                   unsigned int stride_d0,
                                                   unsigned int ds_code) {
    const unsigned long long ga = (unsigned long long)gaddr;
    v4u g0;
    g0[0] = 1u;                                        // count=1, user descriptor
    g0[1] = lds_off;                                   // lds_addr (bytes)
    g0[2] = (unsigned int)ga;                          // global_addr[31:0]
    g0[3] = (unsigned int)(ga >> 32) | (2u << 30);     // global_addr[56:32] | type=2
    v8i g1;
    g1[0] = (int)(ds_code << 16);                      // wg_mask=0 | data_size
    g1[1] = (int)(tile_d0 << 16);                      // tensor_dim0[15:0]
    g1[2] = (int)(tile_d1 << 16);                      // tensor_dim0 hi | tensor_dim1 lo
    g1[3] = (int)(tile_d0 << 16);                      // tensor_dim1 hi | tile_dim0
    g1[4] = (int)tile_d1;                              // tile_dim1 | tile_dim2=0
    g1[5] = (int)stride_d0;                            // tensor_dim0_stride[31:0]
    g1[6] = 0;                                         // stride hi | dim1_stride lo
    g1[7] = 0;
    const v4i z4 = {0, 0, 0, 0};
    const v8i z8 = {0, 0, 0, 0, 0, 0, 0, 0};
    __builtin_amdgcn_tensor_load_to_lds(g0, g1, z4, z4, z8, 0);
}
#endif

// ---------------------------------------------------------------------------
// Stage A: projection  Out[r, d] = sum_e X[r, e] * W[d, e]   (X:[B*S,E], W:[D,E])
// transposed==0 -> Out bf16 [B*S, D] ; transposed==1 -> Out bf16 [B, D, S].
// Block = 8 waves = 128 rows. W is converted to bf16 in LDS once per block;
// the 128x32 f32 X tile is double-buffered into LDS by the Tensor Data Mover.
// ---------------------------------------------------------------------------
__global__ void proj_wmma_kernel(const float* __restrict__ X,
                                 const float* __restrict__ W,
                                 unsigned short* __restrict__ Out,
                                 int transposed) {
    extern __shared__ unsigned char lds_raw[];
    unsigned int* WL = (unsigned int*)lds_raw;                 // 64 x 512 u32 = 128 KB
    float* XT = (float*)(lds_raw + (size_t)Dn * (En / 2) * 4); // 2 x 128 x 32 f32 = 32 KB

    const int tid  = threadIdx.x;
    const int lane = tid & 31;
    const int wave = tid >> 5;
    const int half = lane >> 4;
    const int lm   = lane & 15;
    const long r0b = (long)blockIdx.x * 128;
    const long r0  = r0b + (long)wave * 16;

    // Stage all of W (f32 -> packed bf16x2) into LDS; reused by 32 K-steps.
    for (int idx = tid; idx < Dn * (En / 2); idx += 256) {
        const float2 f = *(const float2*)(W + (long)idx * 2);
        WL[idx] = pack_bf16x2(f.x, f.y);
    }

#ifdef HAVE_TDM
    if (tid == 0) {                                   // TDM: EXEC-independent DMA
        tdm_load_2d((unsigned int)(uintptr_t)XT, X + r0b * En, 32, 128, En, 2);
        __builtin_amdgcn_s_wait_tensorcnt(0);
    }
#endif
    __syncthreads();

    v8f acc[4] = {};
    for (int kc = 0; kc < En / 32; ++kc) {
        const int kb  = kc * 32;
        const int cur = kc & 1;
#ifdef HAVE_TDM
        if (tid == 0 && kb + 32 < En)                 // prefetch next tile into alt buffer
            tdm_load_2d((unsigned int)(uintptr_t)(XT + (cur ^ 1) * 128 * 32),
                        X + r0b * En + kb + 32, 32, 128, En, 2);
        const float* xt = XT + cur * 128 * 32 + (wave * 16 + lm) * 32;
#endif
        Frag16 a;
#pragma unroll
        for (int v = 0; v < 8; ++v) {
            const int ko = a_koff(v, half);
#ifdef HAVE_TDM
            const float2 f = *(const float2*)(xt + ko);
#else
            const float2 f = *(const float2*)(X + (r0 + lm) * En + kb + ko);
#endif
            a.u[v] = pack_bf16x2(f.x, f.y);
        }
#pragma unroll
        for (int nt = 0; nt < 4; ++nt) {
            Frag16 b;
#pragma unroll
            for (int v = 0; v < 8; ++v)
                b.u[v] = WL[(nt * 16 + lm) * (En / 2) + (kb >> 1) + 8 * half + v];
            acc[nt] = __builtin_amdgcn_wmma_f32_16x16x32_bf16(
                false, a.v, false, b.v, (short)0, acc[nt], false, false);
        }
#ifdef HAVE_TDM
        if (tid == 0 && kb + 32 < En) __builtin_amdgcn_s_wait_tensorcnt(0);
        __syncthreads();                              // next buffer ready / safe to reuse
#endif
    }

    if (!transposed) {
#pragma unroll
        for (int nt = 0; nt < 4; ++nt)
#pragma unroll
            for (int r = 0; r < 8; ++r)
                Out[(r0 + r + 8 * half) * Dn + nt * 16 + lm] = f2bf(acc[nt][r]);
    } else {
        const int bidx  = (int)(r0 >> 11);
        const int sbase = (int)(r0 & (Sn - 1));
#pragma unroll
        for (int nt = 0; nt < 4; ++nt)
#pragma unroll
            for (int r = 0; r < 8; ++r)
                Out[((long)bidx * Dn + nt * 16 + lm) * Sn + sbase + r + 8 * half] =
                    f2bf(acc[nt][r]);
    }
}

// ---------------------------------------------------------------------------
// Fused stage B+C: scores = softmax(mask(Q K^T / 8)) -> d_out, then attention
// = P V computed from the LDS-resident bf16 probability tile (never re-read
// from HBM).  LDS: 16x2048 f32 scores (128KB) + 16x2048 bf16 probs (64KB);
// the score region is recycled for the cross-wave PV reduction.
// ---------------------------------------------------------------------------
__global__ void attn_fused_kernel(const unsigned short* __restrict__ Qb, // [B*S, D] bf16
                                  const unsigned short* __restrict__ Kb, // [B*S, D] bf16
                                  const unsigned short* __restrict__ Vt, // [B, D, S] bf16
                                  const unsigned char*  __restrict__ mask,// [B, S, S]
                                  float* __restrict__ P,                  // [B, S, S]
                                  float* __restrict__ AttnOut) {          // [B, S, D]
    extern __shared__ unsigned char lds_raw[];
    float* scoreF       = (float*)lds_raw;                             // 16*Sn f32
    unsigned short* Pb  = (unsigned short*)(lds_raw + (size_t)16 * Sn * 4); // 16*Sn bf16
    float* part         = scoreF;                                      // recycled

    const int tid  = threadIdx.x;
    const int lane = tid & 31;
    const int wave = tid >> 5;
    const int half = lane >> 4;
    const int lm   = lane & 15;
    const int s0   = blockIdx.x * 16;
    const int b    = blockIdx.y;
    const long rowBase = (long)b * Sn + s0;

    // ---- Phase 1: Q K^T / sqrt(D) + mask, into LDS -------------------------
    Frag16 aq[2];
#pragma unroll
    for (int kc = 0; kc < 2; ++kc)
#pragma unroll
        for (int v = 0; v < 8; ++v) {
            const int k = kc * 32 + a_koff(v, half);
            aq[kc].u[v] = *(const unsigned int*)(Qb + (rowBase + lm) * Dn + k);
        }

    for (int j = wave; j < Sn / 16; j += 8) {
        const int t0 = j * 16;
        v8f acc = {};
#pragma unroll
        for (int kc = 0; kc < 2; ++kc) {
            Frag16 bk;                            // B[k][n] = K[t0+n][k]
#pragma unroll
            for (int v = 0; v < 8; ++v) {
                const int k = kc * 32 + b_koff(v, half);
                bk.u[v] = *(const unsigned int*)(Kb + ((long)b * Sn + t0 + lm) * Dn + k);
            }
            acc = __builtin_amdgcn_wmma_f32_16x16x32_bf16(
                false, aq[kc].v, false, bk.v, (short)0, acc, false, false);
        }
#pragma unroll
        for (int r = 0; r < 8; ++r) {
            const int m = r + 8 * half;
            float sc = acc[r] * 0.125f;
            if (mask[((long)b * Sn + s0 + m) * Sn + t0 + lm]) sc = -__builtin_inff();
            scoreF[m * Sn + t0 + lm] = sc;
        }
    }
    __syncthreads();

    // ---- Phase 2: row softmax, 4 columns per lane (float4 LDS reads, b128
    //      global stores on the dominant 134MB output, uint2 bf16 LDS stores).
#pragma unroll
    for (int rr = 0; rr < 2; ++rr) {
        const int m = wave * 2 + rr;
        const float* row = scoreF + m * Sn;
        float mx = -__builtin_inff();
        for (int t = lane * 4; t < Sn; t += 128) {
            const float4 f = *(const float4*)(row + t);
            mx = fmaxf(mx, fmaxf(fmaxf(f.x, f.y), fmaxf(f.z, f.w)));
        }
#pragma unroll
        for (int off = 16; off > 0; off >>= 1) mx = fmaxf(mx, __shfl_xor(mx, off, 32));
        float sum = 0.f;
        for (int t = lane * 4; t < Sn; t += 128) {
            const float4 f = *(const float4*)(row + t);
            sum += __expf(f.x - mx) + __expf(f.y - mx) +
                   __expf(f.z - mx) + __expf(f.w - mx);
        }
#pragma unroll
        for (int off = 16; off > 0; off >>= 1) sum += __shfl_xor(sum, off, 32);
        const float inv = 1.0f / sum;
        float* dst = P + ((long)b * Sn + s0 + m) * Sn;
        for (int t = lane * 4; t < Sn; t += 128) {
            const float4 f = *(const float4*)(row + t);
            float4 p;
            p.x = __expf(f.x - mx) * inv;
            p.y = __expf(f.y - mx) * inv;
            p.z = __expf(f.z - mx) * inv;
            p.w = __expf(f.w - mx) * inv;
            *(float4*)(dst + t) = p;                       // global_store_b128
            uint2 pk;
            pk.x = pack_bf16x2(p.x, p.y);
            pk.y = pack_bf16x2(p.z, p.w);
            *(uint2*)(Pb + m * Sn + t) = pk;               // ds_store_b64
        }
    }
    __syncthreads();

    // ---- Phase 3: partial P.V, each wave owns a 256-wide K slice -----------
    v8f acc[4] = {};
    const int tbase = wave * (Sn / 8);
    for (int kcc = 0; kcc < (Sn / 8) / 32; ++kcc) {
        const int kb = tbase + kcc * 32;
        Frag16 a;                                  // probs from LDS (bf16 packed)
#pragma unroll
        for (int v = 0; v < 8; ++v)
            a.u[v] = *(const unsigned int*)(Pb + lm * Sn + kb + a_koff(v, half));
#pragma unroll
        for (int nt = 0; nt < 4; ++nt) {
            Frag16 bv;                             // B[k][n] = Vt[b, n0+n, k]
#pragma unroll
            for (int v = 0; v < 8; ++v) {
                const int k = kb + b_koff(v, half);
                bv.u[v] = *(const unsigned int*)(Vt + ((long)b * Dn + nt * 16 + lm) * Sn + k);
            }
            acc[nt] = __builtin_amdgcn_wmma_f32_16x16x32_bf16(
                false, a.v, false, bv.v, (short)0, acc[nt], false, false);
        }
    }

    // ---- Phase 4: cross-wave reduction through LDS, write attention --------
#pragma unroll
    for (int nt = 0; nt < 4; ++nt)
#pragma unroll
        for (int r = 0; r < 8; ++r)
            part[wave * (16 * Dn) + (r + 8 * half) * Dn + nt * 16 + lm] = acc[nt][r];
    __syncthreads();

    for (int g = tid; g < 16 * Dn; g += 256) {
        float ssum = 0.f;
#pragma unroll
        for (int w = 0; w < 8; ++w) ssum += part[w * (16 * Dn) + g];
        AttnOut[((long)b * Sn + s0 + (g >> 6)) * Dn + (g & 63)] = ssum;
    }
}

// ---------------------------------------------------------------------------
extern "C" void kernel_launch(void* const* d_in, const int* in_sizes, int n_in,
                              void* d_out, int out_size, void* d_ws, size_t ws_size,
                              hipStream_t stream) {
    const float* q  = (const float*)d_in[0];
    const float* k  = (const float*)d_in[1];
    const float* v  = (const float*)d_in[2];
    const unsigned char* mask = (const unsigned char*)d_in[3];
    const float* WQ = (const float*)d_in[4];
    const float* WK = (const float*)d_in[5];
    const float* WV = (const float*)d_in[6];

    float* attn   = (float*)d_out;                           // [B, S, D]
    float* scores = attn + (size_t)Bn * Sn * Dn;             // [B, S, S]

    unsigned short* Qb = (unsigned short*)d_ws;              // bf16 [B*S, D]
    unsigned short* Kb = Qb + (size_t)Bn * Sn * Dn;          // bf16 [B*S, D]
    unsigned short* Vt = Kb + (size_t)Bn * Sn * Dn;          // bf16 [B, D, S]

    const dim3 blk(256);
    const dim3 gProj((Bn * Sn) / 128);                       // 128 blocks, 8 waves each
    const size_t ldsProj = (size_t)Dn * (En / 2) * 4 +       // W bf16 stage: 128 KB
                           (size_t)2 * 128 * 32 * 4;         // X double buffer: 32 KB
    proj_wmma_kernel<<<gProj, blk, ldsProj, stream>>>(q, WQ, Qb, 0);
    proj_wmma_kernel<<<gProj, blk, ldsProj, stream>>>(k, WK, Kb, 0);
    proj_wmma_kernel<<<gProj, blk, ldsProj, stream>>>(v, WV, Vt, 1);

    const dim3 gAttn(Sn / 16, Bn);                           // (128, 8)
    const size_t ldsAttn = (size_t)16 * Sn * 4 +             // f32 scores: 128 KB
                           (size_t)16 * Sn * 2;              // bf16 probs:  64 KB
    attn_fused_kernel<<<gAttn, blk, ldsAttn, stream>>>(Qb, Kb, Vt, mask, scores, attn);
}
